// CausalSelfAttention_35734127903302
// MI455X (gfx1250) — matrix operand order repaired and checked by reading, not compiled, not run on hardware
//
#include <hip/hip_runtime.h>

typedef __attribute__((ext_vector_type(16))) __bf16 v16bf;
typedef __attribute__((ext_vector_type(8)))  float  v8f;

union BFrag { uint4 u[2]; v16bf v; };

#define DMODEL 1024

// ---- CDNA5 TDM availability (device pass only; host falls back) -----------
#if defined(__AMDGCN__) && defined(__has_builtin)
#  if __has_builtin(__builtin_amdgcn_tensor_load_to_lds)
#    define USE_TDM 1
#  endif
#endif
#ifndef USE_TDM
#  define USE_TDM 0
#endif

#if USE_TDM
typedef __attribute__((ext_vector_type(4))) unsigned int v4u;
typedef __attribute__((ext_vector_type(4))) int          v4i;
typedef __attribute__((ext_vector_type(8))) int          v8i;

// Issue a 2-D TDM load: tile (tile_w x tile_h) of 2-byte elements from a
// row-major tensor with row stride `stride` (elements) into LDS at lds_off.
// D# layout per CDNA5 ISA 8.3/8.4: group0 = flags/lds/global addr, group1 =
// data_size, tensor dims, tile dims, dim0 stride.
__device__ __forceinline__ void tdm_load_2d(const void* gaddr, unsigned lds_off,
                                            unsigned tensor_w, unsigned tensor_h,
                                            unsigned tile_w, unsigned tile_h,
                                            unsigned stride)
{
    unsigned long long ga = (unsigned long long)(uintptr_t)gaddr;
    v4u g0;
    g0.x = 1u;                                          // count=1, user mode
    g0.y = lds_off;                                     // lds_addr [63:32]
    g0.z = (unsigned)(ga & 0xffffffffu);                // global_addr[31:0]
    g0.w = (unsigned)((ga >> 32) & 0x01ffffffu)         // global_addr[56:32]
         | (2u << 30);                                  // type=2 ("image")
    v8i g1;
    g1[0] = (int)(1u << 16);                            // data_size=1 (2B)
    g1[1] = (int)((tensor_w & 0xffffu) << 16);          // tensor_dim0[15:0]
    g1[2] = (int)((tensor_w >> 16) | ((tensor_h & 0xffffu) << 16));
    g1[3] = (int)((tensor_h >> 16) | (tile_w << 16));   // tile_dim0
    g1[4] = (int)(tile_h & 0xffffu);                    // tile_dim1
    g1[5] = (int)stride;                                // dim0_stride[31:0]
    g1[6] = 0;                                          // stride hi / dim1_stride
    g1[7] = 0;
    v4i z4 = {0, 0, 0, 0};
#if __clang_major__ >= 23
    v8i z8 = {0, 0, 0, 0, 0, 0, 0, 0};
    __builtin_amdgcn_tensor_load_to_lds(g0, g1, z4, z4, z8, 0);
#else
    __builtin_amdgcn_tensor_load_to_lds(g0, g1, z4, z4, 0);
#endif
}

__device__ __forceinline__ void wait_tensorcnt0()
{
#if __has_builtin(__builtin_amdgcn_s_wait_tensorcnt)
    __builtin_amdgcn_s_wait_tensorcnt(0);
#else
    asm volatile("s_wait_tensorcnt 0x0" ::: "memory");
#endif
}
#endif  // USE_TDM

// ---------------------------------------------------------------------------
// fp32 -> bf16 conversion
// ---------------------------------------------------------------------------
__global__ __launch_bounds__(256) void cvt_f32_to_bf16(
    const float* __restrict__ src, __bf16* __restrict__ dst, int n)
{
    int i = blockIdx.x * 256 + threadIdx.x;
    if (i < n) dst[i] = (__bf16)src[i];
}

// ---------------------------------------------------------------------------
// C[M,N] = A[M,K] * W[N,K]^T   (bf16 inputs, fp32 accumulate via WMMA)
// Block tile 128x64, BK=32. 8 waves, each wave: 16 rows x 64 cols (4 tiles).
// Staging via TDM (tensor_load_to_lds) with ping-pong LDS double buffering:
// tile k+1 DMA overlaps tile k WMMAs.  Optional fused RoPE (ropeT>0, N-tile
// == one head), optional transposed bf16 store (transT>0), or fp32 store.
// ---------------------------------------------------------------------------
__global__ __launch_bounds__(256) void gemm_nt_bf16(
    const __bf16* __restrict__ A, const __bf16* __restrict__ W,
    int M, int N, int K,
    __bf16* __restrict__ outB, float* __restrict__ outF,
    int ropeT, int transT)
{
    __shared__ __bf16 As[2][128][32];   // 16 KB (ping-pong)
    __shared__ __bf16 Ws[2][64][32];    //  8 KB

    const int tid  = threadIdx.x;
    const int wv   = tid >> 5;          // wave id 0..7
    const int lane = tid & 31;
    const int hl   = lane >> 4;         // lane half
    const int ln   = lane & 15;
    const int m0   = blockIdx.y * 128;
    const int n0   = blockIdx.x * 64;

    const v8f zero = {0.f,0.f,0.f,0.f,0.f,0.f,0.f,0.f};
    v8f acc[4];
#pragma unroll
    for (int i = 0; i < 4; ++i) acc[i] = zero;

    const int nK = K >> 5;

#if USE_TDM
    const unsigned ldsA0 = (unsigned)(uintptr_t)&As[0][0][0];
    const unsigned ldsA1 = (unsigned)(uintptr_t)&As[1][0][0];
    const unsigned ldsW0 = (unsigned)(uintptr_t)&Ws[0][0][0];
    const unsigned ldsW1 = (unsigned)(uintptr_t)&Ws[1][0][0];

    if (wv == 0) {
        tdm_load_2d(A + (size_t)m0 * K, ldsA0, (unsigned)K, (unsigned)M,
                    32u, 128u, (unsigned)K);
        tdm_load_2d(W + (size_t)n0 * K, ldsW0, (unsigned)K, (unsigned)N,
                    32u, 64u, (unsigned)K);
    }
    for (int kt = 0; kt < nK; ++kt) {
        const int bb = kt & 1;
        if (wv == 0) wait_tensorcnt0();      // buffer bb is resident
        __syncthreads();
        if (wv == 0 && kt + 1 < nK) {        // stream next tile while computing
            const int k1 = (kt + 1) << 5;
            tdm_load_2d(A + (size_t)m0 * K + k1, (kt & 1) ? ldsA0 : ldsA1,
                        (unsigned)K, (unsigned)M, 32u, 128u, (unsigned)K);
            tdm_load_2d(W + (size_t)n0 * K + k1, (kt & 1) ? ldsW0 : ldsW1,
                        (unsigned)K, (unsigned)N, 32u, 64u, (unsigned)K);
        }
#else
    for (int kt = 0; kt < nK; ++kt) {
        const int bb = kt & 1;
        const int k0 = kt << 5;
        for (int c = tid; c < 512; c += 256) {
            int r = c >> 2, col = (c & 3) << 3;
            *(uint4*)&As[bb][r][col] =
                *(const uint4*)&A[(size_t)(m0 + r) * K + k0 + col];
        }
        {
            int r = tid >> 2, col = (tid & 3) << 3;
            *(uint4*)&Ws[bb][r][col] =
                *(const uint4*)&W[(size_t)(n0 + r) * K + k0 + col];
        }
        __syncthreads();
#endif
        // A fragment (16x32): lane(m=ln, half=hl) runs K=hl*8.. , 16+hl*8..
        BFrag af;
        af.u[0] = *(const uint4*)&As[bb][wv * 16 + ln][hl * 8];
        af.u[1] = *(const uint4*)&As[bb][wv * 16 + ln][16 + hl * 8];
#pragma unroll
        for (int nb = 0; nb < 4; ++nb) {
            // B fragment (32x16): lane(n=ln, half=hl) runs K=hl*8.. , 16+hl*8..
            BFrag bf;
            bf.u[0] = *(const uint4*)&Ws[bb][nb * 16 + ln][hl * 8];
            bf.u[1] = *(const uint4*)&Ws[bb][nb * 16 + ln][16 + hl * 8];
            acc[nb] = __builtin_amdgcn_wmma_f32_16x16x32_bf16(
                false, af.v, false, bf.v, (short)0, acc[nb], false, false);
        }
        __syncthreads();
    }

    const int mBase = m0 + wv * 16 + hl * 8;  // row(r) = mBase + r

    if (ropeT > 0) {
        // N-tile (64 wide) == one head; pair (i, i+32) is acc[nb] / acc[nb+2]
#pragma unroll
        for (int nb = 0; nb < 2; ++nb) {
            float fi   = (float)(nb * 16 + ln);                 // i in [0,32)
            float invf = __expf(fi * (-9.210340372f / 32.0f));  // theta^(-i/32)
#pragma unroll
            for (int r = 0; r < 8; ++r) {
                float pos = (float)((mBase + r) % ropeT);
                float s, c;
                __sincosf(pos * invf, &s, &c);
                float x1 = acc[nb][r], x2 = acc[nb + 2][r];
                acc[nb][r]     = x1 * c - x2 * s;
                acc[nb + 2][r] = x1 * s + x2 * c;
            }
        }
    }

#pragma unroll
    for (int nb = 0; nb < 4; ++nb) {
        int n = n0 + nb * 16 + ln;
#pragma unroll
        for (int r = 0; r < 8; ++r) {
            int row = mBase + r;
            float v = acc[nb][r];
            if (outF) {
                outF[(size_t)row * N + n] = v;
            } else if (transT > 0) {
                int b = row / transT, t = row - b * transT;
                outB[((size_t)b * N + n) * transT + t] = (__bf16)v;
            } else {
                outB[(size_t)row * N + n] = (__bf16)v;
            }
        }
    }
}

// ---------------------------------------------------------------------------
// Flash attention: one wave per 16-query tile per (b,h). 32-key tiles.
// Q,K bf16 [b*T+t][1024]; V transposed bf16 [b*1024 + dim][T]; O bf16.
// ---------------------------------------------------------------------------
__global__ __launch_bounds__(32) void flash_attn16(
    const __bf16* __restrict__ Q, const __bf16* __restrict__ Kb,
    const __bf16* __restrict__ Vt, __bf16* __restrict__ O,
    int H, int T)
{
    __shared__ __bf16 Ps[16][32];   // P tile for lane-transpose, 1 KB

    const int lane = threadIdx.x;
    const int hl = lane >> 4, ln = lane & 15;
    const int b  = blockIdx.y / H, h = blockIdx.y % H;
    const int q0 = blockIdx.x * 16;

    // Q fragments for 16x64: two 16x32 A-frags (dim halves)
    const __bf16* qbase = Q + ((size_t)(b * T + q0) + ln) * DMODEL + h * 64;
    BFrag qa[2];
#pragma unroll
    for (int f = 0; f < 2; ++f) {
        const __bf16* p = qbase + f * 32 + hl * 8;
        qa[f].u[0] = *(const uint4*)p;
        qa[f].u[1] = *(const uint4*)(p + 16);
    }

    const v8f zero = {0.f,0.f,0.f,0.f,0.f,0.f,0.f,0.f};
    v8f oacc[4];
#pragma unroll
    for (int i = 0; i < 4; ++i) oacc[i] = zero;
    float mrow[8], lrow[8];
#pragma unroll
    for (int r = 0; r < 8; ++r) { mrow[r] = -__builtin_inff(); lrow[r] = 0.f; }

    const int nkt = (q0 + 15) / 32 + 1;   // causal: key tiles up to diagonal
    for (int kt = 0; kt < nkt; ++kt) {
        const int j0 = kt * 32;

        // prefetch next key/value tile into cache while computing this one
        if (kt + 1 < nkt) {
            const __bf16* nk =
                Kb + ((size_t)(b * T + j0 + 32 + lane)) * DMODEL + h * 64;
            __builtin_prefetch(nk, 0, 1);
            const __bf16* nv =
                Vt + ((size_t)(b * DMODEL) + h * 64 + lane * 2) * T + j0 + 32;
            __builtin_prefetch(nv, 0, 1);
        }

        // S = Q K^T for 32 keys: two 16x16 C tiles, chained over 64-dim
        v8f sc[2];
#pragma unroll
        for (int g = 0; g < 2; ++g) {
            const __bf16* kbase =
                Kb + ((size_t)(b * T + j0 + g * 16 + ln)) * DMODEL + h * 64;
            v8f z = zero;
#pragma unroll
            for (int f = 0; f < 2; ++f) {
                BFrag kf;
                const __bf16* p = kbase + f * 32 + hl * 8;
                kf.u[0] = *(const uint4*)p;
                kf.u[1] = *(const uint4*)(p + 16);
                z = __builtin_amdgcn_wmma_f32_16x16x32_bf16(
                    false, qa[f].v, false, kf.v, (short)0, z, false, false);
            }
            sc[g] = z;
        }

        // online softmax (row = r + 8*hl spread over 16 lanes of the half)
        float p0[8], p1[8];
#pragma unroll
        for (int r = 0; r < 8; ++r) {
            int qr = q0 + hl * 8 + r;
            float s0 = sc[0][r] * 0.125f;          // 1/sqrt(64)
            float s1 = sc[1][r] * 0.125f;
            if (j0 + ln      > qr) s0 = -3.0e38f;  // causal mask
            if (j0 + 16 + ln > qr) s1 = -3.0e38f;
            float mx = fmaxf(s0, s1);
#pragma unroll
            for (int off = 1; off < 16; off <<= 1)
                mx = fmaxf(mx, __shfl_xor(mx, off, 32));
            float mnew = fmaxf(mrow[r], mx);
            float corr = __expf(mrow[r] - mnew);
            float e0 = __expf(s0 - mnew);
            float e1 = __expf(s1 - mnew);
            float rs = e0 + e1;
#pragma unroll
            for (int off = 1; off < 16; off <<= 1)
                rs += __shfl_xor(rs, off, 32);
            lrow[r] = lrow[r] * corr + rs;
            mrow[r] = mnew;
            p0[r] = e0; p1[r] = e1;
#pragma unroll
            for (int nb = 0; nb < 4; ++nb) oacc[nb][r] *= corr;
        }

        // transpose P (C-layout -> A-layout) through LDS
#pragma unroll
        for (int r = 0; r < 8; ++r) {
            Ps[hl * 8 + r][ln]      = (__bf16)p0[r];
            Ps[hl * 8 + r][16 + ln] = (__bf16)p1[r];
        }
        // single wave: LDS ops in-order; compiler inserts s_wait_dscnt
        BFrag pf;
        pf.u[0] = *(const uint4*)&Ps[ln][hl * 8];
        pf.u[1] = *(const uint4*)&Ps[ln][16 + hl * 8];

        // O += P V : V^T layout gives contiguous keys per (dim) lane
#pragma unroll
        for (int nb = 0; nb < 4; ++nb) {
            BFrag vf;
            const __bf16* p = Vt +
                ((size_t)(b * DMODEL) + h * 64 + nb * 16 + ln) * T + j0 + hl * 8;
            vf.u[0] = *(const uint4*)p;
            vf.u[1] = *(const uint4*)(p + 16);
            oacc[nb] = __builtin_amdgcn_wmma_f32_16x16x32_bf16(
                false, pf.v, false, vf.v, (short)0, oacc[nb], false, false);
        }
    }

#pragma unroll
    for (int r = 0; r < 8; ++r) {
        float inv = 1.0f / lrow[r];
        size_t row = (size_t)(b * T + q0 + hl * 8 + r);
#pragma unroll
        for (int nb = 0; nb < 4; ++nb) {
            O[row * DMODEL + h * 64 + nb * 16 + ln] =
                (__bf16)(oacc[nb][r] * inv);
        }
    }
}

// ---------------------------------------------------------------------------
// launch
// ---------------------------------------------------------------------------
extern "C" void kernel_launch(void* const* d_in, const int* in_sizes, int n_in,
                              void* d_out, int out_size, void* d_ws, size_t ws_size,
                              hipStream_t stream)
{
    const int B = 4, T = 2048, D = 1024, H = 16;
    const size_t nTok = (size_t)B * T;   // 8192
    const size_t nX   = nTok * D;        // 8,388,608
    const size_t nW   = (size_t)D * D;   // 1,048,576

    const float* x  = (const float*)d_in[0];
    const float* wq = (const float*)d_in[1];
    const float* wk = (const float*)d_in[2];
    const float* wv = (const float*)d_in[3];
    const float* wo = (const float*)d_in[4];

    char* w = (char*)d_ws;
    auto carve = [&](size_t elems) {
        __bf16* p = (__bf16*)w;
        w += elems * sizeof(__bf16);
        return p;
    };
    __bf16* xb  = carve(nX);   // bf16 activations
    __bf16* wqB = carve(nW);
    __bf16* wkB = carve(nW);
    __bf16* wvB = carve(nW);
    __bf16* woB = carve(nW);
    __bf16* qB  = carve(nX);   // RoPE'd Q
    __bf16* kB  = carve(nX);   // RoPE'd K
    __bf16* vT  = carve(nX);   // V transposed [b][dim][t]
    __bf16* aB  = carve(nX);   // attention output

    cvt_f32_to_bf16<<<(int)((nX + 255) / 256), 256, 0, stream>>>(x,  xb,  (int)nX);
    cvt_f32_to_bf16<<<(int)((nW + 255) / 256), 256, 0, stream>>>(wq, wqB, (int)nW);
    cvt_f32_to_bf16<<<(int)((nW + 255) / 256), 256, 0, stream>>>(wk, wkB, (int)nW);
    cvt_f32_to_bf16<<<(int)((nW + 255) / 256), 256, 0, stream>>>(wv, wvB, (int)nW);
    cvt_f32_to_bf16<<<(int)((nW + 255) / 256), 256, 0, stream>>>(wo, woB, (int)nW);

    dim3 gg(D / 64, (int)(nTok / 128));  // (16, 64)
    // Q, K with fused RoPE; V stored transposed for the PV wmma
    gemm_nt_bf16<<<gg, 256, 0, stream>>>(xb, wqB, (int)nTok, D, D, qB, nullptr, T, 0);
    gemm_nt_bf16<<<gg, 256, 0, stream>>>(xb, wkB, (int)nTok, D, D, kB, nullptr, T, 0);
    gemm_nt_bf16<<<gg, 256, 0, stream>>>(xb, wvB, (int)nTok, D, D, vT, nullptr, 0, T);

    flash_attn16<<<dim3(T / 16, B * H), 32, 0, stream>>>(qB, kB, vT, aB, H, T);

    gemm_nt_bf16<<<gg, 256, 0, stream>>>(aB, woB, (int)nTok, D, D,
                                         nullptr, (float*)d_out, 0, 0);
}
